// GNNEncoder_31559419691876
// MI455X (gfx1250) — matrix operand order repaired and checked by reading, not compiled, not run on hardware
//
#include <hip/hip_runtime.h>

typedef _Float16 half_t;
typedef __attribute__((ext_vector_type(16))) _Float16 v16h;
typedef __attribute__((ext_vector_type(2)))  _Float16 v2h;
typedef __attribute__((ext_vector_type(8)))  float    v8f;

#define Kn     12
#define NPER   256
#define BATCH  128
#define NNODES (BATCH * NPER)      // 32768
#define NEDGES (NNODES * Kn)       // 393216
#define MSG_TILES (NEDGES / 16)    // 24576
#define UPD_TILES (NNODES / 16)    // 2048

union H8 { uint4 u; half_t h[8]; };

__device__ __forceinline__ void lds_wave_fence() {
    // CDNA5 split dependency counter: drain outstanding LDS ops (intra-wave
    // store->load ordering) and stop the compiler reordering across it.
    asm volatile("s_wait_dscnt 0" ::: "memory");
}

__device__ __forceinline__ float silu_f(float x) {
    return x / (1.0f + __expf(-x));
}

// A fragment: 16x32 f16 tile, row-major in LDS with leading dim `ld`.
// ISA 7.12.2: lanes 0-15 hold M=lane, K in {0..7,16..23}; lanes 16-31 K in {8..15,24..31}.
// Per-lane elements are two 8-half contiguous runs -> compiler merges to ds_load_b128.
__device__ __forceinline__ v16h load_A(const half_t* p, int ld, int lane) {
    v16h a;
    int m  = lane & 15;
    int kb = (lane >> 4) << 3;           // 0 or 8
#pragma unroll
    for (int j = 0; j < 8; ++j) {
        int k = ((j < 4) ? 0 : 16) + kb + 2 * (j & 3);
        v2h t = *(const v2h*)(p + m * ld + k);
        a[2 * j]     = t.x;
        a[2 * j + 1] = t.y;
    }
    return a;
}

// B fragment: 32x16 f16 tile from N-major (transposed) weights Wt[n][k], ld = K-stride.
// lane = column N; lanes 0-15 hold K in {0..7,16..23}, lanes 16-31 K in {8..15,24..31}.
__device__ __forceinline__ v16h load_B(const half_t* wt, int ld, int lane) {
    v16h b;
    int n  = lane & 15;
    int kb = (lane >> 4) << 3;           // 0 or 8
#pragma unroll
    for (int j = 0; j < 8; ++j) {
        int k = ((j < 4) ? 0 : 16) + kb + 2 * (j & 3);
        v2h t = *(const v2h*)(wt + n * ld + k);
        b[2 * j]     = t.x;
        b[2 * j + 1] = t.y;
    }
    return b;
}

// ---------------------------------------------------------------------------
// Kernel 1: node feature init from species embedding (f32 + f16 shadow copy)
// ---------------------------------------------------------------------------
__global__ void __launch_bounds__(256) init_nodes_kernel(
    const int* __restrict__ species, const float* __restrict__ emb,
    float* __restrict__ nf, half_t* __restrict__ nfh)
{
    int idx  = blockIdx.x * 256 + threadIdx.x;      // < N*64
    int node = idx >> 6, c = idx & 63;
    float v  = emb[species[node] * 64 + c];
    nf[idx]  = v;
    nfh[idx] = (half_t)v;
}

// ---------------------------------------------------------------------------
// Kernel 2: periodic kNN graph build. One block per crystal, thread = atom.
// ---------------------------------------------------------------------------
__global__ void __launch_bounds__(256) graph_kernel(
    const float* __restrict__ lattice, const float* __restrict__ fracs,
    int* __restrict__ edge_src, float* __restrict__ edge_dist)
{
    __shared__ float fx[NPER], fy[NPER], fz[NPER];
    __shared__ float L[9];
    int b = blockIdx.x, i = threadIdx.x;
    const float* f = fracs + (size_t)b * NPER * 3;
    fx[i] = f[i * 3 + 0];
    fy[i] = f[i * 3 + 1];
    fz[i] = f[i * 3 + 2];
    if (i < 9) L[i] = lattice[b * 9 + i];
    __syncthreads();

    float bd[Kn]; int bi[Kn];
#pragma unroll
    for (int k = 0; k < Kn; ++k) { bd[k] = 3.0e38f; bi[k] = 0; }

    float xi = fx[i], yi = fy[i], zi = fz[i];
    for (int j = 0; j < NPER; ++j) {
        if (j == i) continue;
        float dx = xi - fx[j]; dx -= rintf(dx);   // min-image, RNE like jnp.round
        float dy = yi - fy[j]; dy -= rintf(dy);
        float dz = zi - fz[j]; dz -= rintf(dz);
        float cx = dx * L[0] + dy * L[3] + dz * L[6];
        float cy = dx * L[1] + dy * L[4] + dz * L[7];
        float cz = dx * L[2] + dy * L[5] + dz * L[8];
        float d  = sqrtf(cx * cx + cy * cy + cz * cz);
        if (d < bd[Kn - 1]) {                     // sorted insert (single bubble)
            bd[Kn - 1] = d; bi[Kn - 1] = j;
#pragma unroll
            for (int t = Kn - 1; t > 0; --t) {
                if (bd[t] < bd[t - 1]) {
                    float td = bd[t]; bd[t] = bd[t - 1]; bd[t - 1] = td;
                    int   ti = bi[t]; bi[t] = bi[t - 1]; bi[t - 1] = ti;
                }
            }
        }
    }
    size_t eb = ((size_t)b * NPER + i) * Kn;
#pragma unroll
    for (int k = 0; k < Kn; ++k) {
        edge_src[eb + k]  = b * NPER + bi[k];
        edge_dist[eb + k] = bd[k];
    }
}

// ---------------------------------------------------------------------------
// Kernel 3: edge message MLP + scatter-add.  WMMA f32<=f16. Grid-stride over
// 16-edge tiles (weights staged to LDS once per block, amortized 6x).
// ---------------------------------------------------------------------------
__global__ void __launch_bounds__(128) msg_kernel(
    const half_t* __restrict__ nfh,
    const int*    __restrict__ edge_src,
    const float*  __restrict__ edge_dist,
    const float*  __restrict__ W1, const float* __restrict__ b1,   // [192][64],[64]
    const float*  __restrict__ W2, const float* __restrict__ b2,   // [64][64],[64]
    float* __restrict__ agg)                                       // [N][64]
{
    __shared__ __align__(16) half_t sW1t[64 * 192];   // transposed: [n][k]
    __shared__ __align__(16) half_t sW2t[64 * 64];
    __shared__ float sb1[64], sb2[64];
    __shared__ __align__(16) half_t sMsg[4][16 * 192];
    __shared__ __align__(16) half_t sH[4][16 * 64];

    int tid = threadIdx.x;
    for (int i = tid; i < 192 * 64; i += 128) {
        int k = i >> 6, n = i & 63;
        sW1t[n * 192 + k] = (half_t)W1[i];
    }
    for (int i = tid; i < 64 * 64; i += 128) {
        int k = i >> 6, n = i & 63;
        sW2t[n * 64 + k] = (half_t)W2[i];
    }
    if (tid < 64) { sb1[tid] = b1[tid]; sb2[tid] = b2[tid]; }
    __syncthreads();

    int wave = tid >> 5, lane = tid & 31;
    half_t* msg  = sMsg[wave];
    half_t* hbuf = sH[wave];
    int m0 = (lane >> 4) << 3;
    int nc = lane & 15;

    const float STEP = 8.0f / 63.0f;
    const float COEF = -0.5f / (STEP * STEP);

    for (int tile = blockIdx.x * 4 + wave; tile < MSG_TILES; tile += gridDim.x * 4) {
        int e0 = tile * 16;
        lds_wave_fence();   // previous iteration's LDS reads fully drained

        // ---- stage msg_in = [nf[src] | nf[dst] | smear(dist)] as 16x192 f16 ----
        for (int i = lane; i < 16 * 24; i += 32) {
            int e  = i / 24;
            int c8 = (i % 24) * 8;
            int ge = e0 + e;
            H8 v;
            if (c8 < 64) {
                int s = edge_src[ge];
                v.u = *(const uint4*)(nfh + (size_t)s * 64 + c8);
            } else if (c8 < 128) {
                int d = ge / Kn;                     // dst node (edges dst-grouped)
                v.u = *(const uint4*)(nfh + (size_t)d * 64 + (c8 - 64));
            } else {
                float dist = edge_dist[ge];
#pragma unroll
                for (int t = 0; t < 8; ++t) {
                    float z = dist - (float)(c8 - 128 + t) * STEP;
                    v.h[t] = (half_t)__expf(COEF * z * z);
                }
            }
            *(uint4*)(msg + e * 192 + c8) = v.u;
        }
        lds_wave_fence();

        // ---- GEMM1: [16x192] @ [192x64], 6 K-steps x 4 N-tiles ----
        v8f acc[4] = {};
#pragma unroll
        for (int s = 0; s < 6; ++s) {
            v16h a = load_A(msg + s * 32, 192, lane);
#pragma unroll
            for (int nt = 0; nt < 4; ++nt) {
                v16h bw = load_B(sW1t + (nt * 16) * 192 + s * 32, 192, lane);
                acc[nt] = __builtin_amdgcn_wmma_f32_16x16x32_f16(
                    false, a, false, bw, (short)0, acc[nt], false, false);
            }
        }
#pragma unroll
        for (int nt = 0; nt < 4; ++nt) {
            float bb = sb1[nt * 16 + nc];
#pragma unroll
            for (int r = 0; r < 8; ++r) {
                float x = acc[nt][r] + bb;
                hbuf[(m0 + r) * 64 + nt * 16 + nc] = (half_t)silu_f(x);
            }
        }
        lds_wave_fence();

        // ---- GEMM2: [16x64] @ [64x64] ----
        v8f acc2[4] = {};
#pragma unroll
        for (int s = 0; s < 2; ++s) {
            v16h a = load_A(hbuf + s * 32, 64, lane);
#pragma unroll
            for (int nt = 0; nt < 4; ++nt) {
                v16h bw = load_B(sW2t + (nt * 16) * 64 + s * 32, 64, lane);
                acc2[nt] = __builtin_amdgcn_wmma_f32_16x16x32_f16(
                    false, a, false, bw, (short)0, acc2[nt], false, false);
            }
        }
        // ---- bias + segment-sum over dst, run-length merged f32 atomics ----
#pragma unroll
        for (int nt = 0; nt < 4; ++nt) {
            float bb = sb2[nt * 16 + nc];
            int col  = nt * 16 + nc;
            int  pdst = (e0 + m0) / Kn;
            float pend = acc2[nt][0] + bb;
#pragma unroll
            for (int r = 1; r < 8; ++r) {
                int dst = (e0 + m0 + r) / Kn;
                float v = acc2[nt][r] + bb;
                if (dst == pdst) {
                    pend += v;
                } else {
                    atomicAdd(agg + (size_t)pdst * 64 + col, pend);
                    pdst = dst; pend = v;
                }
            }
            atomicAdd(agg + (size_t)pdst * 64 + col, pend);
        }
    }
}

// ---------------------------------------------------------------------------
// Kernel 4: node update MLP + residual + LayerNorm. Grid-stride 16-node tiles.
// ---------------------------------------------------------------------------
__global__ void __launch_bounds__(128) upd_kernel(
    const float* __restrict__ agg,
    float* __restrict__ nf, half_t* __restrict__ nfh,
    const float* __restrict__ W1, const float* __restrict__ b1,
    const float* __restrict__ W2, const float* __restrict__ b2,
    const float* __restrict__ lng, const float* __restrict__ lnb)
{
    __shared__ __align__(16) half_t sW1t[64 * 64];
    __shared__ __align__(16) half_t sW2t[64 * 64];
    __shared__ float sb1[64], sb2[64], sg[64], sbe[64];
    __shared__ __align__(16) half_t sA[4][16 * 64];
    __shared__ __align__(16) half_t sH[4][16 * 64];
    __shared__ float sU[4][16 * 64];

    int tid = threadIdx.x;
    for (int i = tid; i < 64 * 64; i += 128) {
        int k = i >> 6, n = i & 63;
        sW1t[n * 64 + k] = (half_t)W1[i];
        sW2t[n * 64 + k] = (half_t)W2[i];
    }
    if (tid < 64) { sb1[tid] = b1[tid]; sb2[tid] = b2[tid]; sg[tid] = lng[tid]; sbe[tid] = lnb[tid]; }
    __syncthreads();

    int wave = tid >> 5, lane = tid & 31;
    half_t* aT = sA[wave];
    half_t* hb = sH[wave];
    float*  ub = sU[wave];
    int m0 = (lane >> 4) << 3;
    int nc = lane & 15;

    for (int tile = blockIdx.x * 4 + wave; tile < UPD_TILES; tile += gridDim.x * 4) {
        int node0 = tile * 16;
        lds_wave_fence();

        // stage agg tile, f32 -> f16
        for (int i = lane; i < 16 * 8; i += 32) {
            int r = i >> 3, c8 = (i & 7) * 8;
            const float* p = agg + (size_t)(node0 + r) * 64 + c8;
            float4 x = *(const float4*)p;
            float4 y = *(const float4*)(p + 4);
            half_t* q = aT + r * 64 + c8;
            q[0] = (half_t)x.x; q[1] = (half_t)x.y; q[2] = (half_t)x.z; q[3] = (half_t)x.w;
            q[4] = (half_t)y.x; q[5] = (half_t)y.y; q[6] = (half_t)y.z; q[7] = (half_t)y.w;
        }
        lds_wave_fence();

        v8f acc[4] = {};
#pragma unroll
        for (int s = 0; s < 2; ++s) {
            v16h a = load_A(aT + s * 32, 64, lane);
#pragma unroll
            for (int nt = 0; nt < 4; ++nt) {
                v16h bw = load_B(sW1t + (nt * 16) * 64 + s * 32, 64, lane);
                acc[nt] = __builtin_amdgcn_wmma_f32_16x16x32_f16(
                    false, a, false, bw, (short)0, acc[nt], false, false);
            }
        }
#pragma unroll
        for (int nt = 0; nt < 4; ++nt) {
            float bb = sb1[nt * 16 + nc];
#pragma unroll
            for (int r = 0; r < 8; ++r)
                hb[(m0 + r) * 64 + nt * 16 + nc] = (half_t)silu_f(acc[nt][r] + bb);
        }
        lds_wave_fence();

        v8f acc2[4] = {};
#pragma unroll
        for (int s = 0; s < 2; ++s) {
            v16h a = load_A(hb + s * 32, 64, lane);
#pragma unroll
            for (int nt = 0; nt < 4; ++nt) {
                v16h bw = load_B(sW2t + (nt * 16) * 64 + s * 32, 64, lane);
                acc2[nt] = __builtin_amdgcn_wmma_f32_16x16x32_f16(
                    false, a, false, bw, (short)0, acc2[nt], false, false);
            }
        }
        // residual: u + nf -> LDS f32
#pragma unroll
        for (int nt = 0; nt < 4; ++nt) {
            float bb = sb2[nt * 16 + nc];
#pragma unroll
            for (int r = 0; r < 8; ++r) {
                int node = node0 + m0 + r;
                float x = acc2[nt][r] + bb + nf[(size_t)node * 64 + nt * 16 + nc];
                ub[(m0 + r) * 64 + nt * 16 + nc] = x;
            }
        }
        lds_wave_fence();

        // LayerNorm: lanes 0..15 each normalize one row
        if (lane < 16) {
            int r = lane, node = node0 + r;
            float mu = 0.f;
            for (int c = 0; c < 64; ++c) mu += ub[r * 64 + c];
            mu *= (1.0f / 64.0f);
            float var = 0.f;
            for (int c = 0; c < 64; ++c) { float d = ub[r * 64 + c] - mu; var += d * d; }
            var *= (1.0f / 64.0f);
            float inv = rsqrtf(var + 1e-5f);
            for (int c = 0; c < 64; ++c) {
                float y = (ub[r * 64 + c] - mu) * inv * sg[c] + sbe[c];
                nf[(size_t)node * 64 + c]  = y;
                nfh[(size_t)node * 64 + c] = (half_t)y;
            }
        }
    }
}

// ---------------------------------------------------------------------------
// Kernel 5: per-graph mean-pool + 2-layer readout MLP. One block per graph.
// ---------------------------------------------------------------------------
__global__ void __launch_bounds__(256) final_kernel(
    const float* __restrict__ nf, const float* __restrict__ lattice,
    const float* __restrict__ W1, const float* __restrict__ b1,   // [73][128]
    const float* __restrict__ W2, const float* __restrict__ b2,   // [128][256]
    float* __restrict__ out)
{
    __shared__ float h0[73];
    __shared__ float hh[128];
    int b = blockIdx.x, tid = threadIdx.x;

    if (tid < 64) {
        float s = 0.f;
        const float* base = nf + (size_t)b * NPER * 64 + tid;
        for (int i = 0; i < NPER; ++i) s += base[(size_t)i * 64];
        h0[tid] = s * (1.0f / NPER);
    } else if (tid < 73) {
        h0[tid] = lattice[b * 9 + (tid - 64)];
    }
    __syncthreads();

    if (tid < 128) {
        float s = b1[tid];
        for (int i = 0; i < 73; ++i) s += h0[i] * W1[i * 128 + tid];
        hh[tid] = silu_f(s);
    }
    __syncthreads();

    {
        float s = b2[tid];
        for (int i = 0; i < 128; ++i) s += hh[i] * W2[i * 256 + tid];
        if (tid < 128) out[(size_t)b * 128 + tid] = s;                         // mu
        else out[(size_t)BATCH * 128 + (size_t)b * 128 + (tid - 128)] = s;     // sigma
    }
}

// ---------------------------------------------------------------------------
extern "C" void kernel_launch(void* const* d_in, const int* in_sizes, int n_in,
                              void* d_out, int out_size, void* d_ws, size_t ws_size,
                              hipStream_t stream) {
    const float* lattice = (const float*)d_in[0];
    const float* fracs   = (const float*)d_in[1];
    const int*   species = (const int*)d_in[2];
    /* d_in[3] batch_indices: layout is repeat(arange(B), 256) -> implicit */
    const float* emb     = (const float*)d_in[4];
    const float* msg_W1  = (const float*)d_in[5];
    const float* msg_b1  = (const float*)d_in[6];
    const float* msg_W2  = (const float*)d_in[7];
    const float* msg_b2  = (const float*)d_in[8];
    const float* upd_W1  = (const float*)d_in[9];
    const float* upd_b1  = (const float*)d_in[10];
    const float* upd_W2  = (const float*)d_in[11];
    const float* upd_b2  = (const float*)d_in[12];
    const float* ln_g    = (const float*)d_in[13];
    const float* ln_b    = (const float*)d_in[14];
    const float* fin_W1  = (const float*)d_in[15];
    const float* fin_b1  = (const float*)d_in[16];
    const float* fin_W2  = (const float*)d_in[17];
    const float* fin_b2  = (const float*)d_in[18];

    char* ws = (char*)d_ws;
    float*  nf    = (float*)ws;   ws += (size_t)NNODES * 64 * 4;   // 8 MB
    float*  agg   = (float*)ws;   ws += (size_t)NNODES * 64 * 4;   // 8 MB
    float*  edist = (float*)ws;   ws += (size_t)NEDGES * 4;        // 1.5 MB
    int*    esrc  = (int*)ws;     ws += (size_t)NEDGES * 4;        // 1.5 MB
    half_t* nfh   = (half_t*)ws;  ws += (size_t)NNODES * 64 * 2;   // 4 MB

    init_nodes_kernel<<<NNODES * 64 / 256, 256, 0, stream>>>(species, emb, nf, nfh);
    graph_kernel<<<BATCH, 256, 0, stream>>>(lattice, fracs, esrc, edist);

    for (int l = 0; l < 3; ++l) {
        hipMemsetAsync(agg, 0, (size_t)NNODES * 64 * 4, stream);
        msg_kernel<<<1024, 128, 0, stream>>>(                // 6 tiles per wave
            nfh, esrc, edist,
            msg_W1 + (size_t)l * 192 * 64, msg_b1 + l * 64,
            msg_W2 + (size_t)l * 64 * 64,  msg_b2 + l * 64, agg);
        upd_kernel<<<256, 128, 0, stream>>>(                 // 2 tiles per wave
            agg, nf, nfh,
            upd_W1 + (size_t)l * 64 * 64, upd_b1 + l * 64,
            upd_W2 + (size_t)l * 64 * 64, upd_b2 + l * 64,
            ln_g + l * 64, ln_b + l * 64);
    }

    final_kernel<<<BATCH, 256, 0, stream>>>(nf, lattice, fin_W1, fin_b1, fin_W2, fin_b2,
                                            (float*)d_out);
}
